// TimeCaps_16234976379167
// MI455X (gfx1250) — matrix operand order repaired and checked by reading, not compiled, not run on hardware
//
#include <hip/hip_runtime.h>

// ---- dims ----
#define L 1024
#define KCH 64
#define BATCH 64
#define NCLS 10
#define DCAT 8320            // 8*1024 + 128
#define KBLK 260             // 8320 / 32
#define ROWS 512             // BATCH * 8 capsule rows
#define LT 256               // l-tile for fused GEMM kernel
#define XTS 72               // xT row stride in halfwords (64 + 8 pad)

typedef __attribute__((ext_vector_type(16))) __bf16 v16bf;
typedef __attribute__((ext_vector_type(8)))  float  v8f;

union Frag16 { v16bf v; unsigned int u[8]; };

__device__ __forceinline__ unsigned short f2bf(float f) {
    unsigned int u = __float_as_uint(f);
    u += 0x7FFFu + ((u >> 16) & 1u);       // round-to-nearest-even
    return (unsigned short)(u >> 16);
}

// ---------------- K1: conv1  X[64,1,1024] -> x[64,64,1024] ----------------
__global__ void k_conv1(const float* __restrict__ X, const float* __restrict__ W,
                        float* __restrict__ x) {
    int idx = blockIdx.x * blockDim.x + threadIdx.x;   // 64*64*1024
    int l = idx & 1023, o = (idx >> 10) & 63, n = idx >> 16;
    const float* xr = X + (size_t)n * L;
    const float* wr = W + o * 9;
    float acc = 0.f;
#pragma unroll
    for (int t = 0; t < 9; ++t) {
        int p = l + t - 4;
        if (p >= 0 && p < L) acc += xr[p] * wr[t];
    }
    x[idx] = acc;
}

// ---- K1b: pre-swizzle psi_m / convB weights into A-fragment layout -------
// A frag (16-bit 16x32): lane(h,m) reg v holds K = 16*(v>=4) + 8h + 2*(v&3) (+1)
__global__ void k_swzPM(const float* __restrict__ Wm, const float* __restrict__ Wb,
                        unsigned int* __restrict__ wmswz, unsigned int* __restrict__ wbswz) {
    int idx = blockIdx.x * blockDim.x + threadIdx.x;
    if (idx >= 1536) return;
    int lane = idx & 31, half = lane >> 4, m = lane & 15;
    if (idx < 1280) {                              // psi_m: [t][s][mtile][lane]
        int mtile = (idx >> 5) & 3, s = (idx >> 7) & 1, t = idx >> 8;
        int o = mtile * 16 + m;
        unsigned int* dst = wmswz + ((size_t)idx << 3);
#pragma unroll
        for (int v = 0; v < 8; ++v) {
            int kk = ((v >> 2) << 4) + 8 * half + ((v & 3) << 1);
            int i = s * 32 + kk;
            unsigned short lo = f2bf(Wm[o * 320 + i * 5 + t]);
            unsigned short hi = f2bf(Wm[o * 320 + (i + 1) * 5 + t]);
            dst[v] = (unsigned int)lo | ((unsigned int)hi << 16);
        }
    } else {                                       // convB: [s][mtile][lane]
        int j = idx - 1280;
        int mtile = (j >> 5) & 3, s = j >> 7;
        int o = mtile * 16 + m;
        unsigned int* dst = wbswz + ((size_t)j << 3);
#pragma unroll
        for (int v = 0; v < 8; ++v) {
            int kk = ((v >> 2) << 4) + 8 * half + ((v & 3) << 1);
            int i = s * 32 + kk;
            unsigned short lo = f2bf(Wb[o * 64 + i]);
            unsigned short hi = f2bf(Wb[o * 64 + i + 1]);
            dst[v] = (unsigned int)lo | ((unsigned int)hi << 16);
        }
    }
}

// ---- K2: fused WMMA implicit-GEMM: psi_m conv (5 taps) + convB (1x1) -----
// grid = 64 n * 4 l-tiles, 256 threads (8 waves). Wave w owns n-tiles 2w,2w+1.
// mt / t are kept as rolled loops so only one accumulator pair is live and
// fragment loads cannot be LICM-hoisted en masse (avoids scratch spills).
__global__ void __launch_bounds__(256, 1)
k_gemmAB(const float* __restrict__ x,
         const unsigned int* __restrict__ wmswz,
         const unsigned int* __restrict__ wbswz,
         const float* __restrict__ bB,
         float* __restrict__ xa, float* __restrict__ xb1,
         float* __restrict__ sqpart) {
    __shared__ unsigned short xT[260 * XTS];       // transposed bf16 tile [l][i]
    __shared__ float part[8][64];
    int tid = threadIdx.x;
    int n = blockIdx.x >> 2, lt = blockIdx.x & 3;
    int l0 = lt * LT;
    const float* xn = x + (size_t)n * KCH * L;
    for (int e = tid; e < 64 * 260; e += 256) {
        int i = e / 260, lc = e - i * 260;
        int gl = l0 - 2 + lc;
        float v = (gl >= 0 && gl < L) ? xn[i * L + gl] : 0.f;
        xT[lc * XTS + i] = f2bf(v);
    }
    int lane = tid & 31, wid = tid >> 5;
    int half = lane >> 4, ncol = lane & 15;
    for (int o = lane; o < 64; o += 32) part[wid][o] = 0.f;
    __syncthreads();

    const unsigned int* xTu = (const unsigned int*)xT;
    const v8f vz = {0.f, 0.f, 0.f, 0.f, 0.f, 0.f, 0.f, 0.f};
#pragma unroll 1
    for (int ni = 0; ni < 2; ++ni) {
        int nt = wid * 2 + ni;
        int lcol = l0 + nt * 16 + ncol;
#pragma unroll 1
        for (int mt = 0; mt < 4; ++mt) {
            v8f cm = vz, cb = vz;
#pragma unroll 1
            for (int t = 0; t < 5; ++t) {
                int lrow = nt * 16 + ncol + t;
#pragma unroll
                for (int s = 0; s < 2; ++s) {
                    Frag16 B;
                    int ub = lrow * (XTS / 2) + s * 16 + half * 4;
#pragma unroll
                    for (int v = 0; v < 8; ++v) B.u[v] = xTu[ub + ((v >> 2) << 3) + (v & 3)];
                    Frag16 A;
                    const unsigned int* a8 =
                        wmswz + ((size_t)(((t * 2 + s) * 4 + mt) * 32 + lane) << 3);
#pragma unroll
                    for (int v = 0; v < 8; ++v) A.u[v] = a8[v];
                    cm = __builtin_amdgcn_wmma_f32_16x16x32_bf16(
                        false, A.v, false, B.v, (short)0, cm, false, false);
                    if (t == 2) {                  // convB shares the t=2 B fragment
                        Frag16 A2;
                        const unsigned int* a9 =
                            wbswz + ((size_t)((s * 4 + mt) * 32 + lane) << 3);
#pragma unroll
                        for (int v = 0; v < 8; ++v) A2.u[v] = a9[v];
                        cb = __builtin_amdgcn_wmma_f32_16x16x32_bf16(
                            false, A2.v, false, B.v, (short)0, cb, false, false);
                    }
                }
            }
            // epilogue for this M-tile: write xa / xb1, row sum-of-squares
#pragma unroll
            for (int r = 0; r < 8; ++r) {
                int o = mt * 16 + r + 8 * half;
                float val = cm[r];
                xa[((size_t)n * 64 + o) * L + lcol] = val;
                float ssv = val * val;
#pragma unroll
                for (int m = 1; m < 16; m <<= 1) ssv += __shfl_xor(ssv, m);
                if (ncol == 0) part[wid][o] += ssv;
                xb1[((size_t)n * 64 + o) * L + lcol] = cb[r] + bB[o];
            }
        }
    }
    __syncthreads();
    if (tid < 64) {
        float s = 0.f;
#pragma unroll
        for (int w = 0; w < 8; ++w) s += part[w][tid];
        sqpart[((size_t)n * 4 + lt) * 64 + tid] = s;
    }
}

// -------- K3: Cell_A fused: squash-scale + psi_mA conv + routing ----------
__global__ void k_cellA(const float* __restrict__ xa, const float* __restrict__ sqpart,
                        const float* __restrict__ wA, const float* __restrict__ bA,
                        const float* __restrict__ alpha_p,
                        unsigned short* __restrict__ xcat) {
    __shared__ float xa2[8 * 1024];
    __shared__ float wsh[24];
    __shared__ float scaleRow[64];
    __shared__ float bsh[8], bacc[8], wred[8];
    int tid = threadIdx.x;                       // 256
    int n = blockIdx.x >> 6, c = blockIdx.x & 63;
    if (tid < 24) wsh[tid] = wA[c * 24 + tid];
    if (tid < 8) { bsh[tid] = 0.f; bacc[tid] = 0.f; }
    if (tid < 64) {                              // squash scale per input row
        float sq = 0.f;
#pragma unroll
        for (int t = 0; t < 4; ++t) sq += sqpart[((size_t)n * 4 + t) * 64 + tid];
        scaleRow[tid] = sq / ((1.f + sq) * (sq + 1e-8f));
    }
    float bias = bA[c];
    __syncthreads();
    const float* xsn = xa + (size_t)n * KCH * L;
    for (int e = tid; e < 8192; e += 256) {
        int j = e >> 10, l = e & 1023;
        float acc = bias;
#pragma unroll
        for (int p = 0; p < 8; ++p) {
            const float* row = xsn + (j * 8 + p) * L;
            float sR = scaleRow[j * 8 + p];
#pragma unroll
            for (int t = 0; t < 3; ++t) {
                int ll = l + t - 1;
                if (ll >= 0 && ll < L) acc += row[ll] * (sR * wsh[p * 3 + t]);
            }
        }
        xa2[e] = acc;
    }
    __syncthreads();
    float alpha = alpha_p[0];
    int lane = tid & 31, wid = tid >> 5;
    for (int it = 0; it < 2; ++it) {
        float bm = bsh[0];
#pragma unroll
        for (int j = 1; j < 8; ++j) bm = fmaxf(bm, bsh[j]);
        float cc[8], se = 0.f;
#pragma unroll
        for (int j = 0; j < 8; ++j) { cc[j] = __expf(bsh[j] - bm); se += cc[j]; }
        float inv = 1.f / se;
        float sv[4], ss = 0.f;
#pragma unroll
        for (int q = 0; q < 4; ++q) {
            int l = tid + 256 * q;
            float s = 0.f;
#pragma unroll
            for (int j = 0; j < 8; ++j) s += cc[j] * inv * xa2[j * 1024 + l];
            sv[q] = s; ss += s * s;
        }
#pragma unroll
        for (int m = 16; m; m >>= 1) ss += __shfl_xor(ss, m);
        if (lane == 0) wred[wid] = ss;
        __syncthreads();
        float sq = 0.f;
#pragma unroll
        for (int w = 0; w < 8; ++w) sq += wred[w];
        float scale = sq / ((1.f + sq) * (sq + 1e-8f));
        if (it == 0) {
            float pd[8];
#pragma unroll
            for (int j = 0; j < 8; ++j) pd[j] = 0.f;
#pragma unroll
            for (int q = 0; q < 4; ++q) {
                int l = tid + 256 * q;
#pragma unroll
                for (int j = 0; j < 8; ++j) pd[j] += xa2[j * 1024 + l] * sv[q];
            }
#pragma unroll
            for (int j = 0; j < 8; ++j) {
                float v = pd[j];
#pragma unroll
                for (int m = 16; m; m >>= 1) v += __shfl_xor(v, m);
                if (lane == 0) atomicAdd(&bacc[j], v * scale);
            }
            __syncthreads();
            if (tid < 8) { bsh[tid] += bacc[tid]; bacc[tid] = 0.f; }
            __syncthreads();
        } else {
            int c1 = c >> 3, c2 = c & 7;
            unsigned short* dst = xcat + (size_t)(n * 8 + c1) * DCAT + c2 * 1024;
#pragma unroll
            for (int q = 0; q < 4; ++q) {
                int l = tid + 256 * q;
                dst[l] = f2bf(alpha * scale * sv[q]);
            }
        }
    }
}

// -------- K5: Cell_B fused: psi_kB conv + squash + routing, per (n,o) -----
__global__ void k_cellB(const float* __restrict__ xb1, const float* __restrict__ wKB,
                        const float* __restrict__ bKB, const float* __restrict__ beta_p,
                        unsigned short* __restrict__ xcat) {
    __shared__ float wsh[320];
    __shared__ float xb2[8 * 128];
    __shared__ float wr2[8][4];
    __shared__ float bsh[8], bacc[8], wred[4];
    int tid = threadIdx.x;                     // 128
    int n = blockIdx.x >> 3, o = blockIdx.x & 7;
    float bias = bKB[o];
    if (tid < 8) { bsh[tid] = 0.f; bacc[tid] = 0.f; }
    for (int i = tid; i < 320; i += 128) wsh[i] = wKB[o * 320 + i];
    __syncthreads();
    const float* xn = xb1 + (size_t)n * KCH * L;
    int lane = tid & 31, wid = tid >> 5;
    float acc[8];
#pragma unroll
    for (int s = 0; s < 8; ++s) {
        float a = bias;
        for (int i = 0; i < 64; ++i) {
            const float* row = xn + i * L + s * 128;
#pragma unroll
            for (int t = 0; t < 5; ++t) {
                int ww = tid + t - 2;
                if (ww >= 0 && ww < 128) a += row[ww] * wsh[i * 5 + t];
            }
        }
        acc[s] = a;
    }
#pragma unroll
    for (int s = 0; s < 8; ++s) {
        float v = acc[s] * acc[s];
#pragma unroll
        for (int m = 16; m; m >>= 1) v += __shfl_xor(v, m);
        if (lane == 0) wr2[s][wid] = v;
    }
    __syncthreads();
#pragma unroll
    for (int s = 0; s < 8; ++s) {
        float sq = wr2[s][0] + wr2[s][1] + wr2[s][2] + wr2[s][3];
        float scale = sq / ((1.f + sq) * (sq + 1e-8f));
        xb2[s * 128 + tid] = acc[s] * scale;
    }
    __syncthreads();
    float beta = beta_p[0];
    for (int it = 0; it < 2; ++it) {
        float bm = bsh[0];
#pragma unroll
        for (int j = 1; j < 8; ++j) bm = fmaxf(bm, bsh[j]);
        float cc[8], se = 0.f;
#pragma unroll
        for (int j = 0; j < 8; ++j) { cc[j] = __expf(bsh[j] - bm); se += cc[j]; }
        float inv = 1.f / se;
        float sj = 0.f;
#pragma unroll
        for (int j = 0; j < 8; ++j) sj += cc[j] * inv * xb2[j * 128 + tid];
        float ss = sj * sj;
#pragma unroll
        for (int m = 16; m; m >>= 1) ss += __shfl_xor(ss, m);
        if (lane == 0) wred[wid] = ss;
        __syncthreads();
        float sq = wred[0] + wred[1] + wred[2] + wred[3];
        float scale = sq / ((1.f + sq) * (sq + 1e-8f));
        if (it == 0) {
#pragma unroll
            for (int j = 0; j < 8; ++j) {
                float pd = xb2[j * 128 + tid] * sj * scale;
#pragma unroll
                for (int m = 16; m; m >>= 1) pd += __shfl_xor(pd, m);
                if (lane == 0) atomicAdd(&bacc[j], pd);
            }
            __syncthreads();
            if (tid < 8) { bsh[tid] += bacc[tid]; bacc[tid] = 0.f; }
            __syncthreads();
        } else {
            xcat[(size_t)(n * 8 + o) * DCAT + 8192 + tid] = f2bf(beta * scale * sj);
        }
    }
}

// -------- K6: pre-swizzle W_digit to bf16 fragment-major B layout ---------
__global__ void k_swzW(const float* __restrict__ Wd, unsigned int* __restrict__ wswz) {
    int idx = blockIdx.x * blockDim.x + threadIdx.x;
    if (idx >= NCLS * KBLK * 32) return;
    int lane = idx & 31;
    int kb   = (idx >> 5) % KBLK;
    int k    = idx / (KBLK * 32);
    int half = lane >> 4, ncol = lane & 15;
    const float* base = Wd + (size_t)k * DCAT * 16;
    unsigned int* dst = wswz + ((size_t)idx << 3);
#pragma unroll
    for (int v = 0; v < 8; ++v) {
        int kk = kb * 32 + ((v >> 2) << 4) + 8 * half + ((v & 3) << 1);
        unsigned short lo = f2bf(base[(size_t)kk * 16 + ncol]);
        unsigned short hi = f2bf(base[(size_t)(kk + 1) * 16 + ncol]);
        dst[v] = (unsigned int)lo | ((unsigned int)hi << 16);
    }
}

// -------- K7: DigitCaps GEMM (bf16 WMMA, K=8320) + fused routing ----------
__global__ void __launch_bounds__(256, 1)
k_digitcaps(const unsigned short* __restrict__ xcat,
            const unsigned int* __restrict__ wswz,
            float* __restrict__ out) {
    int k    = blockIdx.x >> 5;      // class
    int tile = blockIdx.x & 31;      // 16-row tile of 512 capsule rows
    int tid = threadIdx.x, lane = tid & 31, wid = tid >> 5;
    int half = lane >> 4, mcol = lane & 15;
    int row = tile * 16 + mcol;
    const unsigned int* xrow = (const unsigned int*)xcat + ((size_t)row * DCAT >> 1);
    v8f accv = {0.f, 0.f, 0.f, 0.f, 0.f, 0.f, 0.f, 0.f};
    int kb0 = wid * 33;
    int kb1 = (kb0 + 33 < KBLK) ? kb0 + 33 : KBLK;
    for (int kb = kb0; kb < kb1; ++kb) {
        Frag16 A, B;
        int kbase = kb * 32 + 8 * half;
#pragma unroll
        for (int v = 0; v < 8; ++v) {
            int off = ((v >> 2) << 4) + ((v & 3) << 1);
            A.u[v] = xrow[(kbase + off) >> 1];
        }
        const unsigned int* bp = wswz + ((((size_t)k * KBLK + kb) * 32 + lane) << 3);
#pragma unroll
        for (int v = 0; v < 8; ++v) B.u[v] = bp[v];
        __builtin_prefetch(bp + 256, 0, 1);   // next kb's B stream -> global_prefetch_b8
        accv = __builtin_amdgcn_wmma_f32_16x16x32_bf16(
            false, A.v, false, B.v, (short)0, accv, false, false);
    }
    __shared__ float red[8][32][8];
#pragma unroll
    for (int r = 0; r < 8; ++r) red[wid][lane][r] = accv[r];
    __syncthreads();
    if (tid < 32) {
        float u[8];
#pragma unroll
        for (int r = 0; r < 8; ++r) {
            float s = 0.f;
#pragma unroll
            for (int w = 0; w < 8; ++w) s += red[w][lane][r];
            u[r] = s;   // u[capsule r][out-dim mcol] for batch n = tile*2 + half
        }
        float b[8];
#pragma unroll
        for (int j = 0; j < 8; ++j) b[j] = 0.f;
        int n = tile * 2 + half;
        float vout = 0.f;
        for (int it = 0; it < 2; ++it) {
            float bm = b[0];
#pragma unroll
            for (int j = 1; j < 8; ++j) bm = fmaxf(bm, b[j]);
            float cc[8], se = 0.f;
#pragma unroll
            for (int j = 0; j < 8; ++j) { cc[j] = __expf(b[j] - bm); se += cc[j]; }
            float inv = 1.f / se;
            float sj = 0.f;
#pragma unroll
            for (int j = 0; j < 8; ++j) sj += cc[j] * inv * u[j];
            float ss = sj * sj;
#pragma unroll
            for (int m = 1; m < 16; m <<= 1) ss += __shfl_xor(ss, m);
            float scale = ss / ((1.f + ss) * (ss + 1e-8f));
            if (it == 0) {
#pragma unroll
                for (int j = 0; j < 8; ++j) {
                    float pd = u[j] * sj * scale;
#pragma unroll
                    for (int m = 1; m < 16; m <<= 1) pd += __shfl_xor(pd, m);
                    b[j] += pd;
                }
            } else vout = scale * sj;
        }
        out[((size_t)n * NCLS + k) * 16 + mcol] = vout;
    }
}

extern "C" void kernel_launch(void* const* d_in, const int* in_sizes, int n_in,
                              void* d_out, int out_size, void* d_ws, size_t ws_size,
                              hipStream_t stream) {
    const float* X        = (const float*)d_in[0];
    const float* conv1_w  = (const float*)d_in[1];
    const float* psi_m_w  = (const float*)d_in[2];
    const float* psi_mA_w = (const float*)d_in[3];
    const float* psi_mA_b = (const float*)d_in[4];
    const float* convB_w  = (const float*)d_in[5];
    const float* convB_b  = (const float*)d_in[6];
    const float* psi_kB_w = (const float*)d_in[7];
    const float* psi_kB_b = (const float*)d_in[8];
    const float* W_digit  = (const float*)d_in[9];
    const float* alpha    = (const float*)d_in[10];
    const float* beta     = (const float*)d_in[11];

    const size_t NEL = (size_t)BATCH * KCH * L;            // 4,194,304
    float* x      = (float*)d_ws;                          // 16 MB
    float* xa     = x + NEL;                               // 16 MB
    float* xb1    = xa + NEL;                              // 16 MB
    float* sqpart = xb1 + NEL;                             // 64*4*64 = 16384 f
    unsigned short* xcat = (unsigned short*)(sqpart + 16384);        // 8.5 MB
    unsigned int* wswz  = (unsigned int*)(xcat + (size_t)ROWS * DCAT); // 2.66 MB
    unsigned int* wmswz = wswz + (size_t)NCLS * KBLK * 32 * 8;       // 40 KB
    unsigned int* wbswz = wmswz + 1280 * 8;                          // 8 KB

    k_conv1<<<16384, 256, 0, stream>>>(X, conv1_w, x);
    k_swzPM<<<6, 256, 0, stream>>>(psi_m_w, convB_w, wmswz, wbswz);
    k_swzW<<<325, 256, 0, stream>>>(W_digit, wswz);
    k_gemmAB<<<BATCH * 4, 256, 0, stream>>>(x, wmswz, wbswz, convB_b, xa, xb1, sqpart);
    k_cellA<<<4096, 256, 0, stream>>>(xa, sqpart, psi_mA_w, psi_mA_b, alpha, xcat);
    k_cellB<<<512, 128, 0, stream>>>(xb1, psi_kB_w, psi_kB_b, beta, xcat);
    k_digitcaps<<<NCLS * 32, 256, 0, stream>>>(xcat, wswz, (float*)d_out);
}
